// DiffNet_25615184953444
// MI455X (gfx1250) — matrix operand order repaired and checked by reading, not compiled, not run on hardware
//
#include <hip/hip_runtime.h>
#include <hip/hip_bf16.h>
#include <math.h>

// ---------------------------------------------------------------------------
// Problem constants (from reference)
// ---------------------------------------------------------------------------
#define HID   1024
#define INW   128
#define OUTW  32
#define TPRE  64
#define TFWD  128
#define BATCH 128
#define XD    96
#define YD    32
#define G3H   (3 * HID)   // 3072

typedef __attribute__((ext_vector_type(16))) __bf16 v16bf;
typedef __attribute__((ext_vector_type(8)))  __bf16 v8bf;
typedef __attribute__((ext_vector_type(8)))  float  v8f;

// Load a 16x32 bf16 A-fragment: lane l -> row m0+(l&15),
// K slots {16*half+0..15} (contiguous, matches B), two contiguous 16B loads.
__device__ __forceinline__ v16bf load_afrag(const __bf16* __restrict__ arow,
                                            int k0, int half)
{
    v8bf a_lo = *(const v8bf*)(arow + k0 + 16 * half);
    v8bf a_hi = *(const v8bf*)(arow + k0 + 16 * half + 8);
    return __builtin_shufflevector(a_lo, a_hi,
                0,1,2,3,4,5,6,7,8,9,10,11,12,13,14,15);
}

__device__ __forceinline__ v8f wmma_bf16(v16bf a, v16bf b, v8f c)
{
    return __builtin_amdgcn_wmma_f32_16x16x32_bf16(
               false, a, false, b, (short)0, c, false, false);
}

// ---------------------------------------------------------------------------
// Fully fused GRU cell.
// Block = 256 thr (8 waves), grid = HID/16 = 64. Block b owns columns
// j in [16*b, 16*b+16). Wave w owns rows [16w, 16w+16) (M = B = 128).
// Each wave accumulates 6 strips of 16x16: {r,z,n} x {x@Wi.T, h@Wh.T},
// then does bias + sigmoid/tanh + blend in the epilogue and writes
// h (f32, in place, element-owned) and hbf_out (bf16 ping-pong).
// ---------------------------------------------------------------------------
template <int KI>
__global__ __launch_bounds__(256) void k_gru_cell(
    const __bf16* __restrict__ Ai,  const __bf16* __restrict__ Wi,
    const __bf16* __restrict__ Ah,  const __bf16* __restrict__ Wh,
    const float*  __restrict__ bi,  const float*  __restrict__ bh,
    float* __restrict__ h, __bf16* __restrict__ hbf_out)
{
    const int lane = threadIdx.x & 31;
    const int wave = threadIdx.x >> 5;
    const int l15  = lane & 15;
    const int half = lane >> 4;
    const int m0   = wave * 16;
    const int j0   = blockIdx.x * 16;

    v8f ai0 = {}, ai1 = {}, ai2 = {};   // gi gates r,z,n
    v8f ah0 = {}, ah1 = {}, ah2 = {};   // gh gates r,z,n

    // ---- gi strips: x @ Wi.T at columns {j0, H+j0, 2H+j0} ----
    {
        const __bf16* arow = Ai + (size_t)(m0 + l15) * KI;
        const __bf16* w0 = Wi + (size_t)(0 * HID + j0 + l15) * KI + 16 * half;
        const __bf16* w1 = Wi + (size_t)(1 * HID + j0 + l15) * KI + 16 * half;
        const __bf16* w2 = Wi + (size_t)(2 * HID + j0 + l15) * KI + 16 * half;
        for (int k0 = 0; k0 < KI; k0 += 32) {
            v16bf af = load_afrag(arow, k0, half);
            ai0 = wmma_bf16(af, *(const v16bf*)(w0 + k0), ai0);
            ai1 = wmma_bf16(af, *(const v16bf*)(w1 + k0), ai1);
            ai2 = wmma_bf16(af, *(const v16bf*)(w2 + k0), ai2);
        }
    }
    // ---- gh strips: h @ Wh.T (K = HID always) ----
    {
        const __bf16* arow = Ah + (size_t)(m0 + l15) * HID;
        const __bf16* w0 = Wh + (size_t)(0 * HID + j0 + l15) * HID + 16 * half;
        const __bf16* w1 = Wh + (size_t)(1 * HID + j0 + l15) * HID + 16 * half;
        const __bf16* w2 = Wh + (size_t)(2 * HID + j0 + l15) * HID + 16 * half;
        for (int k0 = 0; k0 < HID; k0 += 32) {
            v16bf af = load_afrag(arow, k0, half);
            ah0 = wmma_bf16(af, *(const v16bf*)(w0 + k0), ah0);
            ah1 = wmma_bf16(af, *(const v16bf*)(w1 + k0), ah1);
            ah2 = wmma_bf16(af, *(const v16bf*)(w2 + k0), ah2);
        }
    }

    // ---- GRU epilogue: D layout -> row m0+v+8*half, col j0+l15 ----
    const int j = j0 + l15;
    const float bir = bi[j],           bhr = bh[j];
    const float biz = bi[HID + j],     bhz = bh[HID + j];
    const float bin = bi[2 * HID + j], bhn = bh[2 * HID + j];
#pragma unroll
    for (int v = 0; v < 8; ++v) {
        const int m = m0 + v + 8 * half;
        float r = 1.0f / (1.0f + expf(-(ai0[v] + bir + ah0[v] + bhr)));
        float z = 1.0f / (1.0f + expf(-(ai1[v] + biz + ah1[v] + bhz)));
        float n = tanhf(ai2[v] + bin + r * (ah2[v] + bhn));
        const size_t idx = (size_t)m * HID + j;
        float hnew = (1.0f - z) * n + z * h[idx];
        h[idx]       = hnew;
        hbf_out[idx] = (__bf16)hnew;
    }
}

// ---------------------------------------------------------------------------
// fc1: a1 = tanh(h1 @ fc1.T) -> bf16.  M=128, N=1024, K=1024.
// Block = 8 waves, wave w -> rows [16w,16w+16), block -> 32 cols. Grid = 32.
// ---------------------------------------------------------------------------
__global__ __launch_bounds__(256) void k_fc1_tanh(
    const __bf16* __restrict__ A, const __bf16* __restrict__ W,
    __bf16* __restrict__ Cbf)
{
    const int lane = threadIdx.x & 31;
    const int wave = threadIdx.x >> 5;
    const int l15  = lane & 15;
    const int half = lane >> 4;
    const int m0   = wave * 16;
    const int n0   = blockIdx.x * 32;

    const __bf16* arow = A + (size_t)(m0 + l15) * HID;
    const __bf16* w0 = W + (size_t)(n0 + l15) * HID + 16 * half;
    const __bf16* w1 = W + (size_t)(n0 + 16 + l15) * HID + 16 * half;

    v8f acc0 = {}, acc1 = {};
    for (int k0 = 0; k0 < HID; k0 += 32) {
        v16bf af = load_afrag(arow, k0, half);
        acc0 = wmma_bf16(af, *(const v16bf*)(w0 + k0), acc0);
        acc1 = wmma_bf16(af, *(const v16bf*)(w1 + k0), acc1);
    }
#pragma unroll
    for (int v = 0; v < 8; ++v) {
        const int m = m0 + v + 8 * half;
        Cbf[(size_t)m * HID + n0 + l15]      = (__bf16)tanhf(acc0[v]);
        Cbf[(size_t)m * HID + n0 + 16 + l15] = (__bf16)tanhf(acc1[v]);
    }
}

// ---------------------------------------------------------------------------
// Fused fc2 + feedback: ey = a1 @ fc2.T + last_y ; ys[t] = ey ;
// inp = concat(fx[t] (bf16), ey).  Single block: M=128, N=32, K=1024.
// ---------------------------------------------------------------------------
__global__ __launch_bounds__(256) void k_fc2_ey(
    const __bf16* __restrict__ A, const __bf16* __restrict__ W,
    float* __restrict__ lasty, float* __restrict__ yout,
    const __bf16* __restrict__ fxt, __bf16* __restrict__ inp)
{
    const int lane = threadIdx.x & 31;
    const int wave = threadIdx.x >> 5;
    const int l15  = lane & 15;
    const int half = lane >> 4;
    const int m0   = wave * 16;

    const __bf16* arow = A + (size_t)(m0 + l15) * HID;
    const __bf16* w0 = W + (size_t)(l15) * HID + 16 * half;
    const __bf16* w1 = W + (size_t)(16 + l15) * HID + 16 * half;

    v8f acc0 = {}, acc1 = {};
    for (int k0 = 0; k0 < HID; k0 += 32) {
        v16bf af = load_afrag(arow, k0, half);
        acc0 = wmma_bf16(af, *(const v16bf*)(w0 + k0), acc0);
        acc1 = wmma_bf16(af, *(const v16bf*)(w1 + k0), acc1);
    }
#pragma unroll
    for (int v = 0; v < 8; ++v) {
        const int m = m0 + v + 8 * half;
        {
            const int n = l15;
            float e = acc0[v] + lasty[m * OUTW + n];
            lasty[m * OUTW + n] = e;
            yout[m * OUTW + n]  = e;
            inp[m * INW + XD + n] = (__bf16)e;
        }
        {
            const int n = 16 + l15;
            float e = acc1[v] + lasty[m * OUTW + n];
            lasty[m * OUTW + n] = e;
            yout[m * OUTW + n]  = e;
            inp[m * INW + XD + n] = (__bf16)e;
        }
    }
    // copy forward_x[t] (already bf16) into inp[:, 0:96] (disjoint region)
    for (int i = threadIdx.x; i < BATCH * XD; i += 256) {
        const int b = i / XD, c = i - b * XD;
        inp[b * INW + c] = fxt[i];
    }
}

// ---------------------------------------------------------------------------
// Setup kernels
// ---------------------------------------------------------------------------
__global__ __launch_bounds__(256) void k_f32_to_bf16(
    const float* __restrict__ s, __bf16* __restrict__ d, int n)
{
    int i = blockIdx.x * 256 + threadIdx.x;
    if (i < n) d[i] = (__bf16)s[i];
}

__global__ __launch_bounds__(256) void k_build_preX(
    const float* __restrict__ px, const float* __restrict__ py,
    __bf16* __restrict__ d)
{
    int idx = blockIdx.x * 256 + threadIdx.x;  // TPRE*B*IN
    int c  = idx & (INW - 1);
    int rt = idx >> 7;                         // t*B + b
    float v = (c < XD) ? px[(size_t)rt * XD + c] : py[(size_t)rt * YD + (c - XD)];
    d[idx] = (__bf16)v;
}

__global__ __launch_bounds__(256) void k_zero_state(
    float* __restrict__ h, __bf16* __restrict__ ha, __bf16* __restrict__ hb, int n)
{
    int i = blockIdx.x * 256 + threadIdx.x;
    if (i < n) { h[i] = 0.0f; ha[i] = (__bf16)0.0f; hb[i] = (__bf16)0.0f; }
}

__global__ __launch_bounds__(256) void k_copy_f32(
    const float* __restrict__ s, float* __restrict__ d, int n)
{
    int i = blockIdx.x * 256 + threadIdx.x;
    if (i < n) d[i] = s[i];
}

// ---------------------------------------------------------------------------
// Host orchestration
// ---------------------------------------------------------------------------
extern "C" void kernel_launch(void* const* d_in, const int* in_sizes, int n_in,
                              void* d_out, int out_size, void* d_ws, size_t ws_size,
                              hipStream_t stream)
{
    (void)in_sizes; (void)n_in; (void)out_size; (void)ws_size;

    const float* pre_x = (const float*)d_in[0];
    const float* pre_y = (const float*)d_in[1];
    const float* fwd_x = (const float*)d_in[2];
    const float* Wi0   = (const float*)d_in[3];
    const float* Wh0   = (const float*)d_in[4];
    const float* bi0   = (const float*)d_in[5];
    const float* bh0   = (const float*)d_in[6];
    const float* Wi1   = (const float*)d_in[7];
    const float* Wh1   = (const float*)d_in[8];
    const float* bi1   = (const float*)d_in[9];
    const float* bh1   = (const float*)d_in[10];
    const float* fc1   = (const float*)d_in[11];
    const float* fc2   = (const float*)d_in[12];
    float* ys = (float*)d_out;

    // -------- workspace layout --------
    char*  base = (char*)d_ws;
    size_t off  = 0;
    auto carve = [&](size_t bytes) -> char* {
        char* p = base + off;
        off = (off + bytes + 255) & ~(size_t)255;
        return p;
    };
    __bf16* Wi0b  = (__bf16*)carve((size_t)G3H * INW * 2);
    __bf16* Wh0b  = (__bf16*)carve((size_t)G3H * HID * 2);
    __bf16* Wi1b  = (__bf16*)carve((size_t)G3H * HID * 2);
    __bf16* Wh1b  = (__bf16*)carve((size_t)G3H * HID * 2);
    __bf16* fc1b  = (__bf16*)carve((size_t)HID * HID * 2);
    __bf16* fc2b  = (__bf16*)carve((size_t)OUTW * HID * 2);
    __bf16* preXb = (__bf16*)carve((size_t)TPRE * BATCH * INW * 2);
    __bf16* fxb   = (__bf16*)carve((size_t)TFWD * BATCH * XD * 2);
    __bf16* a1b   = (__bf16*)carve((size_t)BATCH * HID * 2);
    float*  lasty = (float*)carve((size_t)BATCH * OUTW * 4);
    __bf16* inp   = (__bf16*)carve((size_t)BATCH * INW * 2);
    float*  h0    = (float*)carve((size_t)BATCH * HID * 4);
    float*  h1    = (float*)carve((size_t)BATCH * HID * 4);
    __bf16* h0b_a = (__bf16*)carve((size_t)BATCH * HID * 2);
    __bf16* h0b_b = (__bf16*)carve((size_t)BATCH * HID * 2);
    __bf16* h1b_a = (__bf16*)carve((size_t)BATCH * HID * 2);
    __bf16* h1b_b = (__bf16*)carve((size_t)BATCH * HID * 2);

    auto blks = [](int n) { return (n + 255) / 256; };

    // -------- setup: weight / input conversion, state init --------
    k_f32_to_bf16<<<blks(G3H * INW), 256, 0, stream>>>(Wi0, Wi0b, G3H * INW);
    k_f32_to_bf16<<<blks(G3H * HID), 256, 0, stream>>>(Wh0, Wh0b, G3H * HID);
    k_f32_to_bf16<<<blks(G3H * HID), 256, 0, stream>>>(Wi1, Wi1b, G3H * HID);
    k_f32_to_bf16<<<blks(G3H * HID), 256, 0, stream>>>(Wh1, Wh1b, G3H * HID);
    k_f32_to_bf16<<<blks(HID * HID), 256, 0, stream>>>(fc1, fc1b, HID * HID);
    k_f32_to_bf16<<<blks(OUTW * HID), 256, 0, stream>>>(fc2, fc2b, OUTW * HID);
    k_f32_to_bf16<<<blks(TFWD * BATCH * XD), 256, 0, stream>>>(fwd_x, fxb, TFWD * BATCH * XD);
    k_build_preX<<<blks(TPRE * BATCH * INW), 256, 0, stream>>>(pre_x, pre_y, preXb);
    k_zero_state<<<blks(BATCH * HID), 256, 0, stream>>>(h0, h0b_a, h0b_b, BATCH * HID);
    k_zero_state<<<blks(BATCH * HID), 256, 0, stream>>>(h1, h1b_a, h1b_b, BATCH * HID);
    // last_y = pre_y[-1]
    k_copy_f32<<<blks(BATCH * OUTW), 256, 0, stream>>>(
        pre_y + (size_t)(TPRE - 1) * BATCH * YD, lasty, BATCH * OUTW);

    const int CELL_GRID = HID / 16;   // 64 blocks
    const int FC1_GRID  = HID / 32;   // 32 blocks

    __bf16 *h0cur = h0b_a, *h0nxt = h0b_b;
    __bf16 *h1cur = h1b_a, *h1nxt = h1b_b;

    // -------- pre-phase: 64 steps, 2 fused-cell kernels per step --------
    for (int t = 0; t < TPRE; ++t) {
        const __bf16* xt = preXb + (size_t)t * BATCH * INW;
        k_gru_cell<INW><<<CELL_GRID, 256, 0, stream>>>(
            xt, Wi0b, h0cur, Wh0b, bi0, bh0, h0, h0nxt);
        k_gru_cell<HID><<<CELL_GRID, 256, 0, stream>>>(
            h0nxt, Wi1b, h1cur, Wh1b, bi1, bh1, h1, h1nxt);
        { __bf16* tmp = h0cur; h0cur = h0nxt; h0nxt = tmp; }
        { __bf16* tmp = h1cur; h1cur = h1nxt; h1nxt = tmp; }
    }

    // -------- autoregressive phase: 128 steps, 4 kernels per step --------
    for (int t = 0; t < TFWD; ++t) {
        k_fc1_tanh<<<FC1_GRID, 256, 0, stream>>>(h1cur, fc1b, a1b);
        k_fc2_ey<<<1, 256, 0, stream>>>(
            a1b, fc2b, lasty, ys + (size_t)t * BATCH * OUTW,
            fxb + (size_t)t * BATCH * XD, inp);
        k_gru_cell<INW><<<CELL_GRID, 256, 0, stream>>>(
            inp, Wi0b, h0cur, Wh0b, bi0, bh0, h0, h0nxt);
        k_gru_cell<HID><<<CELL_GRID, 256, 0, stream>>>(
            h0nxt, Wi1b, h1cur, Wh1b, bi1, bh1, h1, h1nxt);
        { __bf16* tmp = h0cur; h0cur = h0nxt; h0nxt = tmp; }
        { __bf16* tmp = h1cur; h1cur = h1nxt; h1nxt = tmp; }
    }
}
